// TemporalGraphNetwork_61924838474176
// MI455X (gfx1250) — matrix operand order repaired and checked by reading, not compiled, not run on hardware
//
#include <hip/hip_runtime.h>
#include <stdint.h>

#define N_NODES 2048
#define F_IN 128
#define HID 256
#define HEADS 8
#define NEG_SLOPE 0.2f
#define MASK_VAL -9e15f
#define LN_EPS 1e-5f

typedef __attribute__((ext_vector_type(16))) __bf16 bf16x16;
typedef __attribute__((ext_vector_type(8)))  float  floatx8;
typedef __attribute__((ext_vector_type(4)))  unsigned int u32x4;
typedef __attribute__((ext_vector_type(8)))  int i32x8;
typedef __attribute__((ext_vector_type(4)))  int i32x4;

union Frag {
  unsigned int u[8];
  bf16x16 v;
};

__device__ __forceinline__ unsigned short f2bf(float x) {
  unsigned int b = __float_as_uint(x);
  b += 0x7fffu + ((b >> 16) & 1u);          // round-to-nearest-even
  return (unsigned short)(b >> 16);
}
__device__ __forceinline__ unsigned int f2bf2(float lo, float hi) {
  return (unsigned int)f2bf(lo) | ((unsigned int)f2bf(hi) << 16);
}
__device__ __forceinline__ float bf2f(unsigned short u) {
  return __uint_as_float(((unsigned int)u) << 16);
}
__device__ __forceinline__ float lrelu(float x) { return x >= 0.f ? x : NEG_SLOPE * x; }

__device__ __forceinline__ float block_sum(float v, float* red) {
  const int t = threadIdx.x;
  red[t] = v; __syncthreads();
  for (int s = 128; s > 0; s >>= 1) {
    if (t < s) red[t] += red[t + s];
    __syncthreads();
  }
  float r = red[0]; __syncthreads();
  return r;
}
__device__ __forceinline__ float block_max(float v, float* red) {
  const int t = threadIdx.x;
  red[t] = v; __syncthreads();
  for (int s = 128; s > 0; s >>= 1) {
    if (t < s) red[t] = fmaxf(red[t], red[t + s]);
    __syncthreads();
  }
  float r = red[0]; __syncthreads();
  return r;
}

// ---------------------------------------------------------------------------
// TDM: DMA a 2D tile (32 bf16 in dim0, 256 rows in dim1, row stride 2048 elems)
// from global into LDS, inserting 1 DWORD of padding after every 16 DWORDs so
// each 32-element row lands on a 17-u32 (68 B) stride -> bank-conflict-free.
// Tracked with TENSORcnt; issued by one wave per workgroup.
// This toolchain exposes the 6-arg builtin:
//   (u32x4 g0, i32x8 g1, i32x4 g2, i32x4 g3, i32x8 extra, i32 cpol)
__device__ __forceinline__ void tdm_load_B_tile(const unsigned short* gbase, unsigned lds_off) {
  const unsigned long long ga = (unsigned long long)(uintptr_t)gbase;
  u32x4 g0;
  g0[0] = 1u;                                              // count=1, user descriptor
  g0[1] = lds_off;                                         // lds_addr (bytes)
  g0[2] = (unsigned)ga;                                    // global_addr[31:0]
  g0[3] = (unsigned)((ga >> 32) & 0x01FFFFFFu) | (2u << 30); // global_addr[56:32], type=2
  i32x8 g1;
  g1[0] = (int)((1u << 16) | (1u << 20) | (3u << 22));     // data_size=2B, pad_enable,
                                                           // pad_interval=16 DW, pad_amount=1 DW
  g1[1] = 0;                                               // abar addr=0, tensor_dim0[15:0]
  g1[2] = (int)0x00004000u;                                // tensor_dim0=1<<30, tensor_dim1 lo
  g1[3] = (int)((32u << 16) | 0x4000u);                    // tensor_dim1=1<<30, tile_dim0=32
  g1[4] = (int)256u;                                       // tile_dim1=256, tile_dim2=0
  g1[5] = (int)2048u;                                      // tensor_dim0_stride = 2048 elems
  g1[6] = 0;
  g1[7] = 0;                                               // dim1 stride unused (2D tile)
  const i32x4 z4 = {0, 0, 0, 0};
  const i32x8 z8 = {0, 0, 0, 0, 0, 0, 0, 0};
  __builtin_amdgcn_tensor_load_to_lds(g0, g1, z4, z4, z8, 0);
}

// ---------------------------------------------------------------------------
// f32 [n] -> bf16 [n]
__global__ __launch_bounds__(256)
void cast_bf16_kernel(const float* __restrict__ src, unsigned short* __restrict__ dst, int n) {
  int i = blockIdx.x * 256 + threadIdx.x;
  if (i < n) dst[i] = f2bf(src[i]);
}

// f32 src[R][C] -> bf16 dst[C][R]  (LDS 32x32 tile transpose)
__global__ __launch_bounds__(256)
void transpose_cast_kernel(const float* __restrict__ src, unsigned short* __restrict__ dst,
                           int R, int C) {
  __shared__ float tile[32][33];
  const int tx = threadIdx.x, ty = threadIdx.y;   // 32 x 8
  const int c0 = blockIdx.x * 32;
  const int r0 = blockIdx.y * 32;
  #pragma unroll
  for (int k = 0; k < 32; k += 8)
    tile[ty + k][tx] = src[(size_t)(r0 + ty + k) * C + (c0 + tx)];
  __syncthreads();
  #pragma unroll
  for (int k = 0; k < 32; k += 8)
    dst[(size_t)(c0 + ty + k) * R + (r0 + tx)] = f2bf(tile[tx][ty + k]);
}

// ---------------------------------------------------------------------------
// ht = h @ W, written transposed as bf16:  htT[c][n] = sum_k hb[n][k]*WT[c][k]
// One wave computes a 16(M) x 128(N) f32 tile with 16x16x32 bf16 WMMA.
__global__ __launch_bounds__(32)
void gemm_ht_kernel(const unsigned short* __restrict__ hb,
                    const unsigned short* __restrict__ WT,
                    unsigned short* __restrict__ htT, int K) {
  const int lane = threadIdx.x;
  const int li = lane & 15;
  const int hi = lane >> 4;
  const int Mbase = blockIdx.x * 16;
  const int Nbase = blockIdx.y * 128;

  floatx8 acc[8];
  #pragma unroll
  for (int t = 0; t < 8; ++t)
    #pragma unroll
    for (int k = 0; k < 8; ++k) acc[t][k] = 0.f;

  const unsigned int* hrow = (const unsigned int*)(hb + (size_t)(Mbase + li) * K);

  for (int kb = 0; kb < K; kb += 32) {
    Frag A;
    #pragma unroll
    for (int v = 0; v < 8; ++v) {
      const int u32idx = (kb >> 1) + ((v < 4) ? v : 8 + (v - 4)) + hi * 4;
      A.u[v] = hrow[u32idx];
    }
    #pragma unroll
    for (int t = 0; t < 8; ++t) {
      const unsigned int* wrow = (const unsigned int*)(WT + (size_t)(Nbase + 16 * t + li) * K);
      Frag B;
      #pragma unroll
      for (int v = 0; v < 8; ++v) B.u[v] = wrow[(kb >> 1) + ((v < 4) ? v : 8 + (v - 4)) + hi * 4];
      acc[t] = __builtin_amdgcn_wmma_f32_16x16x32_bf16(
          false, A.v, false, B.v, (short)0, acc[t], false, false);
    }
  }

  #pragma unroll
  for (int t = 0; t < 8; ++t) {
    const int c = Nbase + 16 * t + li;
    uint4 pk;
    pk.x = f2bf2(acc[t][0], acc[t][1]);
    pk.y = f2bf2(acc[t][2], acc[t][3]);
    pk.z = f2bf2(acc[t][4], acc[t][5]);
    pk.w = f2bf2(acc[t][6], acc[t][7]);
    *(uint4*)(htT + (size_t)c * N_NODES + Mbase + 8 * hi) = pk;
  }
}

// ---------------------------------------------------------------------------
// s1[n,h] = <ht[n,h,:], a[:256]> ; s2[n,h] = <ht[n,h,:], a[256:]>
__global__ __launch_bounds__(256)
void s12_kernel(const unsigned short* __restrict__ htT, const float* __restrict__ a,
                float* __restrict__ s1, float* __restrict__ s2) {
  const int n = blockIdx.x * 256 + threadIdx.x;
  for (int h = 0; h < HEADS; ++h) {
    float r1 = 0.f, r2 = 0.f;
    for (int d = 0; d < HID; ++d) {
      const float v = bf2f(htT[(size_t)(h * HID + d) * N_NODES + n]);
      r1 += v * a[d];
      r2 += v * a[HID + d];
    }
    s1[n * HEADS + h] = r1;
    s2[n * HEADS + h] = r2;
  }
}

// ---------------------------------------------------------------------------
// Row softmax stats over j for every (i,h): m = max_j e, invl = 1/sum_j exp(e-m).
__global__ __launch_bounds__(256)
void softmax_stats_kernel(const int* __restrict__ adj, const float* __restrict__ s1,
                          const float* __restrict__ s2, float* __restrict__ mR,
                          float* __restrict__ iL) {
  __shared__ float red[256];
  const int i = blockIdx.x;
  const int t = threadIdx.x;
  int av[8];
  float sv[8][8];
  #pragma unroll
  for (int jj = 0; jj < 8; ++jj) {
    const int j = t + jj * 256;
    av[jj] = adj[(size_t)i * N_NODES + j];
    #pragma unroll
    for (int h = 0; h < HEADS; ++h) sv[jj][h] = s2[j * HEADS + h];
  }
  for (int h = 0; h < HEADS; ++h) {
    const float s1v = s1[i * HEADS + h];
    float mloc = MASK_VAL;
    #pragma unroll
    for (int jj = 0; jj < 8; ++jj) {
      const float e = av[jj] ? lrelu(s1v + sv[jj][h]) : MASK_VAL;
      mloc = fmaxf(mloc, e);
    }
    const float m = block_max(mloc, red);
    float sloc = 0.f;
    #pragma unroll
    for (int jj = 0; jj < 8; ++jj) {
      const float e = av[jj] ? lrelu(s1v + sv[jj][h]) : MASK_VAL;
      sloc += __expf(e - m);
    }
    const float ssum = block_sum(sloc, red);
    if (t == 0) {
      mR[i * HEADS + h] = m;
      iL[i * HEADS + h] = 1.0f / ssum;
    }
  }
}

// ---------------------------------------------------------------------------
// hp[h][i][d] = sum_j att[i,j,h] * ht[j,h,d], att computed on the fly.
// Block = 8 waves, one head; wave w owns 16 rows i. The 32x256 bf16 B slab is
// DMA'd into LDS by the Tensor Data Mover (double-buffered, overlapping WMMA).
__global__ __launch_bounds__(256)
void gat_agg_kernel(const unsigned short* __restrict__ htT, const int* __restrict__ adj,
                    const float* __restrict__ s1, const float* __restrict__ s2,
                    const float* __restrict__ mR, const float* __restrict__ iL,
                    float* __restrict__ hp) {
  __shared__ unsigned int Bsh[2][256 * 17];  // padded rows (17 u32): conflict-free
  __shared__ float s2sh[32];

  const int t = threadIdx.x;
  const int lane = t & 31;
  const int wave = t >> 5;
  const int li = lane & 15;
  const int hi = lane >> 4;
  const int h = blockIdx.y;
  const int ibase = blockIdx.x * 128 + wave * 16;
  const int irow = ibase + li;

  const float s1v = s1[irow * HEADS + h];
  const float mv  = mR[irow * HEADS + h];
  const float il  = iL[irow * HEADS + h];
  const int* adjrow = adj + (size_t)irow * N_NODES;

  floatx8 acc[16];
  #pragma unroll
  for (int dt = 0; dt < 16; ++dt)
    #pragma unroll
    for (int k = 0; k < 8; ++k) acc[dt][k] = 0.f;

  const unsigned short* gtile = htT + (size_t)(h * HID) * N_NODES;  // rows h*256.., col 0
  const unsigned lds_b0 = (unsigned)(uintptr_t)(&Bsh[0][0]);
  const unsigned lds_b1 = (unsigned)(uintptr_t)(&Bsh[1][0]);

  if (wave == 0) tdm_load_B_tile(gtile, lds_b0);   // prologue: tile 0 -> buf 0

  for (int k = 0; k < N_NODES / 32; ++k) {
    if (wave == 0) __builtin_amdgcn_s_wait_tensorcnt(0);
    __syncthreads();                               // tile k resident; buf[(k+1)&1] free
    if (wave == 0 && k + 1 < N_NODES / 32)
      tdm_load_B_tile(gtile + (k + 1) * 32, (k & 1) ? lds_b0 : lds_b1);

    const int jb = k * 32;
    if (t < 32) s2sh[t] = s2[(jb + t) * HEADS + h];
    __syncthreads();

    // A fragment (16x32 attention weights, bf16), computed in registers.
    Frag A;
    #pragma unroll
    for (int v = 0; v < 8; ++v) {
      const int j0 = jb + ((v < 4) ? 2 * v : 16 + 2 * (v - 4)) + hi * 8;
      const int2 a2 = *(const int2*)(adjrow + j0);
      float e0 = (a2.x != 0) ? lrelu(s1v + s2sh[j0 - jb]) : MASK_VAL;
      float e1 = (a2.y != 0) ? lrelu(s1v + s2sh[j0 + 1 - jb]) : MASK_VAL;
      const float p0 = __expf(e0 - mv) * il;
      const float p1 = __expf(e1 - mv) * il;
      A.u[v] = f2bf2(p0, p1);
    }

    const unsigned int* Bs = Bsh[k & 1];
    #pragma unroll
    for (int dt = 0; dt < 16; ++dt) {
      Frag B;
      #pragma unroll
      for (int v = 0; v < 8; ++v)
        B.u[v] = Bs[(dt * 16 + li) * 17 + ((v < 4) ? v : 8 + (v - 4)) + hi * 4];
      acc[dt] = __builtin_amdgcn_wmma_f32_16x16x32_bf16(
          false, A.v, false, B.v, (short)0, acc[dt], false, false);
    }
  }

  #pragma unroll
  for (int dt = 0; dt < 16; ++dt) {
    const int d = dt * 16 + li;
    #pragma unroll
    for (int v = 0; v < 8; ++v) {
      const int i = ibase + 8 * hi + v;
      hp[((size_t)h * N_NODES + i) * HID + d] = acc[dt][v];
    }
  }
}

// ---------------------------------------------------------------------------
// out[n][d] = relu(LN(mean_h hp[h][n][:]))
__global__ __launch_bounds__(256)
void post_kernel(const float* __restrict__ hp, const float* __restrict__ lnw,
                 const float* __restrict__ lnb, float* __restrict__ out) {
  __shared__ float red[256];
  const int n = blockIdx.x;
  const int d = threadIdx.x;
  float v = 0.f;
  #pragma unroll
  for (int h = 0; h < HEADS; ++h)
    v += hp[((size_t)h * N_NODES + n) * HID + d];
  v *= (1.0f / HEADS);
  const float mu = block_sum(v, red) * (1.0f / HID);
  const float dv = v - mu;
  const float var = block_sum(dv * dv, red) * (1.0f / HID);
  const float y = dv * rsqrtf(var + LN_EPS) * lnw[d] + lnb[d];
  out[(size_t)n * HID + d] = fmaxf(y, 0.f);
}

// ---------------------------------------------------------------------------
// node_types = h @ clf_w + clf_b ; anomaly = sigmoid(h @ anom_w + anom_b)
__global__ __launch_bounds__(256)
void heads_kernel(const float* __restrict__ h, const float* __restrict__ clfw,
                  const float* __restrict__ clfb, const float* __restrict__ aw,
                  const float* __restrict__ ab, float* __restrict__ types,
                  float* __restrict__ anom) {
  const int n = blockIdx.x * 256 + threadIdx.x;
  const float* hr = h + (size_t)n * HID;
  float sc[10];
  #pragma unroll
  for (int c = 0; c < 10; ++c) sc[c] = clfb[c];
  float sa = ab[0];
  for (int d = 0; d < HID; ++d) {
    const float x = hr[d];
    #pragma unroll
    for (int c = 0; c < 10; ++c) sc[c] += x * clfw[d * 10 + c];
    sa += x * aw[d];
  }
  #pragma unroll
  for (int c = 0; c < 10; ++c) types[n * 10 + c] = sc[c];
  anom[n] = 1.0f / (1.0f + __expf(-sa));
}

// ---------------------------------------------------------------------------
extern "C" void kernel_launch(void* const* d_in, const int* in_sizes, int n_in,
                              void* d_out, int out_size, void* d_ws, size_t ws_size,
                              hipStream_t stream) {
  const float* node_features = (const float*)d_in[0];
  const float* clf_w  = (const float*)d_in[13];
  const float* clf_b  = (const float*)d_in[14];
  const float* anom_w = (const float*)d_in[15];
  const float* anom_b = (const float*)d_in[16];
  const int*   adj    = (const int*)d_in[17];

  char* wp = (char*)d_ws;
  auto take = [&](size_t bytes) -> void* {
    void* p = (void*)wp;
    wp += (bytes + 255) & ~(size_t)255;
    return p;
  };
  unsigned short* hb  = (unsigned short*)take((size_t)N_NODES * HID * 2);
  unsigned short* WT  = (unsigned short*)take((size_t)N_NODES * HID * 2);
  unsigned short* htT = (unsigned short*)take((size_t)N_NODES * N_NODES * 2);
  float* s1 = (float*)take((size_t)N_NODES * HEADS * 4);
  float* s2 = (float*)take((size_t)N_NODES * HEADS * 4);
  float* mR = (float*)take((size_t)N_NODES * HEADS * 4);
  float* iL = (float*)take((size_t)N_NODES * HEADS * 4);
  float* hp = (float*)take((size_t)HEADS * N_NODES * HID * 4);
  float* hbuf = (float*)take((size_t)N_NODES * HID * 4);

  float* h_final = (float*)d_out;
  float* types   = h_final + (size_t)N_NODES * HID;
  float* anom    = types + (size_t)N_NODES * 10;

  const float* hcur = node_features;
  for (int L = 0; L < 3; ++L) {
    const float* W   = (const float*)d_in[1 + 4 * L];
    const float* a   = (const float*)d_in[2 + 4 * L];
    const float* lnw = (const float*)d_in[3 + 4 * L];
    const float* lnb = (const float*)d_in[4 + 4 * L];
    const int K = (L == 0) ? F_IN : HID;

    transpose_cast_kernel<<<dim3(N_NODES / 32, K / 32), dim3(32, 8), 0, stream>>>(W, WT, K, N_NODES);
    cast_bf16_kernel<<<(N_NODES * K + 255) / 256, 256, 0, stream>>>(hcur, hb, N_NODES * K);
    gemm_ht_kernel<<<dim3(N_NODES / 16, N_NODES / 128), 32, 0, stream>>>(hb, WT, htT, K);
    s12_kernel<<<N_NODES / 256, 256, 0, stream>>>(htT, a, s1, s2);
    softmax_stats_kernel<<<N_NODES, 256, 0, stream>>>(adj, s1, s2, mR, iL);
    gat_agg_kernel<<<dim3(N_NODES / 128, HEADS), 256, 0, stream>>>(htT, adj, s1, s2, mR, iL, hp);
    float* outh = (L == 2) ? h_final : hbuf;
    post_kernel<<<N_NODES, 256, 0, stream>>>(hp, lnw, lnb, outh);
    hcur = outh;
  }
  heads_kernel<<<N_NODES / 256, 256, 0, stream>>>(h_final, clf_w, clf_b, anom_w, anom_b, types, anom);
}
